// PhysNetDemo_31791347925872
// MI455X (gfx1250) — matrix operand order repaired and checked by reading, not compiled, not run on hardware
//
#include <hip/hip_runtime.h>
#include <math.h>

#define N_ATOMS 50000
#define N_EDGES 800000
#define N_MOL   2000
#define F_DIM   128
#define K_RBF   64
#define O_DIM   2
#define RC_F    10.0f
#define LN2_F   0.6931472f

typedef __bf16 bf16;
typedef __attribute__((ext_vector_type(16))) __bf16 v16bf;
typedef __attribute__((ext_vector_type(8)))  __bf16 v8bf;
typedef __attribute__((ext_vector_type(8)))  float  v8f;

union AFrag { v16bf v; v8bf h[2]; };

__device__ __forceinline__ float sspf(float x) {
  // softplus(x) - ln2, numerically stable
  float ax = fabsf(x);
  return fmaxf(x, 0.f) + log1pf(__expf(-ax)) - LN2_F;
}

// ---------------------------------------------------------------- utilities
__global__ __launch_bounds__(256) void k_zero(float* __restrict__ p, int n) {
  int i = blockIdx.x * 256 + threadIdx.x;
  if (i < n) p[i] = 0.f;
}

__global__ __launch_bounds__(256)
void k_transpose_bf16(const float* __restrict__ src, bf16* __restrict__ dst,
                      int rows, int cols) {
  int i = blockIdx.x * 256 + threadIdx.x;
  if (i < rows * cols) {
    int r = i / cols, c = i - r * cols;
    dst[c * rows + r] = (bf16)src[i];
  }
}

__global__ __launch_bounds__(256)
void k_gather_embed(const int* __restrict__ Z, const float* __restrict__ embed,
                    float* __restrict__ x) {
  int i = blockIdx.x * 256 + threadIdx.x;
  if (i < N_ATOMS * F_DIM) {
    int a = i >> 7, j = i & 127;
    x[i] = embed[Z[a] * F_DIM + j];
  }
}

__global__ __launch_bounds__(256)
void k_ssp_bf16(const float* __restrict__ in, bf16* __restrict__ out, int n) {
  int i = blockIdx.x * 256 + threadIdx.x;
  if (i < n) out[i] = (bf16)sspf(in[i]);
}

// ---------------------------------------------------------------- RBF edges
__global__ __launch_bounds__(256)
void k_rbf(const float* __restrict__ R, const int* __restrict__ esrc,
           const int* __restrict__ edst, const float* __restrict__ centers,
           const float* __restrict__ widths, bf16* __restrict__ rbfb) {
  __shared__ float sc[K_RBF], sw[K_RBF];
  if (threadIdx.x < K_RBF) {
    sc[threadIdx.x] = centers[threadIdx.x];
    sw[threadIdx.x] = widths[threadIdx.x];
  }
  __syncthreads();
  int e = blockIdx.x * 256 + threadIdx.x;   // grid sized exactly
  int s = esrc[e], d = edst[e];
  float dx = R[s*3+0] - R[d*3+0];
  float dy = R[s*3+1] - R[d*3+1];
  float dz = R[s*3+2] - R[d*3+2];
  float dist = sqrtf(dx*dx + dy*dy + dz*dz + 1e-12f);
  float t = dist / RC_F;
  float t2 = t*t, t3 = t2*t, t4 = t3*t, t5 = t4*t;
  float phi = (dist < RC_F) ? (1.f - 6.f*t5 + 15.f*t4 - 10.f*t3) : 0.f;
  float ed = __expf(-dist);
  bf16* out = rbfb + (size_t)e * K_RBF;
  #pragma unroll 8
  for (int k = 0; k < K_RBF; ++k) {
    float df = ed - sc[k];
    out[k] = (bf16)(__expf(-sw[k] * df * df) * phi);
  }
}

// ---------------------------------------------------------------- WMMA GEMM
// Y[nRows,128] = act(A_bf16[nRows,128] @ W[128,128]) + b (+ epilogue)
// Wt is n-major bf16 (Wt[n*128+k]). 8 waves/block, wave handles 16 rows x 128 cols.
// MODE: 0 = f32 out; 1 = ssp -> bf16 out; 2 = resid + f32 out; 3 = u*resid + f32 out
template <int MODE>
__global__ __launch_bounds__(256)
void k_gemm(const bf16* __restrict__ A, const bf16* __restrict__ Wt,
            const float* __restrict__ bias, const float* __restrict__ resid,
            const float* __restrict__ uvec, float* __restrict__ outF,
            bf16* __restrict__ outB, int nRows) {
  __shared__ alignas(32) bf16 sW[F_DIM * F_DIM];   // 32 KB
  {
    const uint4* s = (const uint4*)Wt;
    uint4* d = (uint4*)sW;
    #pragma unroll
    for (int i = 0; i < 8; ++i) d[threadIdx.x + i * 256] = s[threadIdx.x + i * 256];
  }
  __syncthreads();

  const int wave = threadIdx.x >> 5;
  const int lane = threadIdx.x & 31;
  const int nlo  = lane & 15;
  const int kh   = lane >> 4;
  const int rowBase = blockIdx.x * 128 + wave * 16;

  // A fragments: lane holds row (rowBase+nlo); K contiguous: k = 16*(lane>>4) + s
  int arow = rowBase + nlo; if (arow > nRows - 1) arow = nRows - 1;
  const bf16* ap = A + (size_t)arow * F_DIM;
  AFrag a[4];
  #pragma unroll
  for (int kb = 0; kb < 4; ++kb) {
    a[kb].h[0] = *(const v8bf*)(ap + kb * 32 + kh * 16);
    a[kb].h[1] = *(const v8bf*)(ap + kb * 32 + kh * 16 + 8);
  }

  #pragma unroll
  for (int nt = 0; nt < 8; ++nt) {
    const int col = nt * 16 + nlo;
    v8f acc = {0.f, 0.f, 0.f, 0.f, 0.f, 0.f, 0.f, 0.f};
    #pragma unroll
    for (int kb = 0; kb < 4; ++kb) {
      v16bf b = *(const v16bf*)(sW + col * F_DIM + kb * 32 + kh * 16);
      acc = __builtin_amdgcn_wmma_f32_16x16x32_bf16(false, a[kb].v, false, b,
                                                    (short)0, acc, false, false);
    }
    const float bcol = bias[col];
    #pragma unroll
    for (int v = 0; v < 8; ++v) {
      const int row = rowBase + v + 8 * kh;   // D layout: M = v + 8*(lane>>4)
      if (row < nRows) {
        float val = acc[v] + bcol;
        size_t idx = (size_t)row * F_DIM + col;
        if (MODE == 0)      outF[idx] = val;
        else if (MODE == 1) outB[idx] = (bf16)sspf(val);
        else if (MODE == 2) outF[idx] = resid[idx] + val;
        else                outF[idx] = uvec[col] * resid[idx] + val;
      }
    }
  }
}

// ------------------------------------------------- edge GEMM + gather/scatter
// g = rbf[16 edges,64] @ G[64,128] (WMMA); msg = g * tj[src]; v[dst] += msg
__global__ __launch_bounds__(256)
void k_edge(const bf16* __restrict__ rbfb, const bf16* __restrict__ Gt,
            const float* __restrict__ tj, float* __restrict__ vbuf,
            const int* __restrict__ esrc, const int* __restrict__ edst) {
  __shared__ alignas(32) bf16 sG[F_DIM * K_RBF];   // 16 KB, n-major
  {
    const uint4* s = (const uint4*)Gt;
    uint4* d = (uint4*)sG;
    #pragma unroll
    for (int i = 0; i < 4; ++i) d[threadIdx.x + i * 256] = s[threadIdx.x + i * 256];
  }
  __syncthreads();

  const int wave = threadIdx.x >> 5;
  const int lane = threadIdx.x & 31;
  const int nlo  = lane & 15;
  const int kh   = lane >> 4;
  const int e0   = (blockIdx.x * 8 + wave) * 16;   // grid sized exactly

  AFrag a[2];
  const bf16* ap = rbfb + (size_t)(e0 + nlo) * K_RBF;
  #pragma unroll
  for (int kb = 0; kb < 2; ++kb) {
    a[kb].h[0] = *(const v8bf*)(ap + kb * 32 + kh * 16);
    a[kb].h[1] = *(const v8bf*)(ap + kb * 32 + kh * 16 + 8);
  }

  int sarr[8], darr[8];
  #pragma unroll
  for (int v = 0; v < 8; ++v) {
    int e = e0 + v + 8 * kh;         // edge for D-matrix row M = v + 8*kh
    sarr[v] = esrc[e];
    darr[v] = edst[e];
    __builtin_prefetch(&tj[(size_t)sarr[v] * F_DIM], 0, 0);
  }

  #pragma unroll
  for (int nt = 0; nt < 8; ++nt) {
    const int col = nt * 16 + nlo;
    v8f acc = {0.f, 0.f, 0.f, 0.f, 0.f, 0.f, 0.f, 0.f};
    #pragma unroll
    for (int kb = 0; kb < 2; ++kb) {
      v16bf b = *(const v16bf*)(sG + col * K_RBF + kb * 32 + kh * 16);
      acc = __builtin_amdgcn_wmma_f32_16x16x32_bf16(false, a[kb].v, false, b,
                                                    (short)0, acc, false, false);
    }
    #pragma unroll
    for (int v = 0; v < 8; ++v) {
      float msg = acc[v] * tj[(size_t)sarr[v] * F_DIM + col];
      (void)__hip_atomic_fetch_add(&vbuf[(size_t)darr[v] * F_DIM + col], msg,
                                   __ATOMIC_RELAXED, __HIP_MEMORY_SCOPE_AGENT);
    }
  }
}

// ---------------------------------------------------- output head (128 -> 2)
__global__ __launch_bounds__(256)
void k_out_acc(const float* __restrict__ o, const float* __restrict__ Wout,
               const float* __restrict__ bout, float* __restrict__ out_sum) {
  const int wave = threadIdx.x >> 5, lane = threadIdx.x & 31;
  const int atom = blockIdx.x * 8 + wave;   // grid sized exactly (50000/8)
  const float* op = o + (size_t)atom * F_DIM;
  float s0 = 0.f, s1 = 0.f;
  #pragma unroll
  for (int k = lane; k < F_DIM; k += 32) {
    float sv = sspf(op[k]);
    s0 += sv * Wout[k * 2 + 0];
    s1 += sv * Wout[k * 2 + 1];
  }
  #pragma unroll
  for (int off = 16; off > 0; off >>= 1) {
    s0 += __shfl_xor(s0, off, 32);
    s1 += __shfl_xor(s1, off, 32);
  }
  if (lane == 0) {
    out_sum[atom * 2 + 0] += s0 + bout[0];
    out_sum[atom * 2 + 1] += s1 + bout[1];
  }
}

// ---------------------------------------------------- molecule segment sums
__global__ __launch_bounds__(256)
void k_final(const float* __restrict__ out_sum, const int* __restrict__ Z,
             const float* __restrict__ scale, const float* __restrict__ shift,
             const int* __restrict__ amb, const float* __restrict__ R,
             float* __restrict__ out) {
  int i = blockIdx.x * 256 + threadIdx.x;
  if (i >= N_ATOMS) return;
  int z = Z[i];
  float a0 = scale[z * 2 + 0] * out_sum[i * 2 + 0] + shift[z * 2 + 0];
  float a1 = scale[z * 2 + 1] * out_sum[i * 2 + 1] + shift[z * 2 + 1];
  int mol = amb[i];
  float* base = out + (size_t)mol * 5;
  (void)__hip_atomic_fetch_add(base + 0, a0, __ATOMIC_RELAXED, __HIP_MEMORY_SCOPE_AGENT);
  (void)__hip_atomic_fetch_add(base + 1, a1, __ATOMIC_RELAXED, __HIP_MEMORY_SCOPE_AGENT);
  (void)__hip_atomic_fetch_add(base + 2, a1 * R[i*3+0], __ATOMIC_RELAXED, __HIP_MEMORY_SCOPE_AGENT);
  (void)__hip_atomic_fetch_add(base + 3, a1 * R[i*3+1], __ATOMIC_RELAXED, __HIP_MEMORY_SCOPE_AGENT);
  (void)__hip_atomic_fetch_add(base + 4, a1 * R[i*3+2], __ATOMIC_RELAXED, __HIP_MEMORY_SCOPE_AGENT);
}

// ======================================================================
extern "C" void kernel_launch(void* const* d_in, const int* in_sizes, int n_in,
                              void* d_out, int out_size, void* d_ws, size_t ws_size,
                              hipStream_t stream) {
  const float* R       = (const float*)d_in[0];
  const int*   Z       = (const int*)d_in[1];
  const int*   esrc    = (const int*)d_in[2];
  const int*   edst    = esrc + N_EDGES;
  const int*   amb     = (const int*)d_in[3];
  const float* embed   = (const float*)d_in[4];
  const float* centers = (const float*)d_in[5];
  const float* widths  = (const float*)d_in[6];
  const float* Wi      = (const float*)d_in[7];
  const float* bi      = (const float*)d_in[8];
  const float* Wj      = (const float*)d_in[9];
  const float* bj      = (const float*)d_in[10];
  const float* G       = (const float*)d_in[11];
  const float* u       = (const float*)d_in[12];
  const float* Wv      = (const float*)d_in[13];
  const float* bv      = (const float*)d_in[14];
  const float* resW    = (const float*)d_in[15];
  const float* resb    = (const float*)d_in[16];
  const float* Wout    = (const float*)d_in[17];
  const float* bout    = (const float*)d_in[18];
  const float* scale   = (const float*)d_in[19];
  const float* shift   = (const float*)d_in[20];
  float* out = (float*)d_out;

  // ---- workspace carve (deterministic)
  size_t off = 0;
  char* base = (char*)d_ws;
  auto take = [&](size_t bytes) -> char* {
    char* p = base + off;
    off += (bytes + 255) & ~(size_t)255;
    return p;
  };
  bf16*  wT   = (bf16*)take(27ull * F_DIM * F_DIM * sizeof(bf16));  // 9 per m, n-major
  bf16*  gT   = (bf16*)take(3ull * F_DIM * K_RBF * sizeof(bf16));
  bf16*  rbfb = (bf16*)take((size_t)N_EDGES * K_RBF * sizeof(bf16));
  float* x    = (float*)take((size_t)N_ATOMS * F_DIM * sizeof(float));
  float* vb   = (float*)take((size_t)N_ATOMS * F_DIM * sizeof(float));
  float* tjb  = (float*)take((size_t)N_ATOMS * F_DIM * sizeof(float));
  float* ob   = (float*)take((size_t)N_ATOMS * F_DIM * sizeof(float));
  bf16*  sxb  = (bf16*)take((size_t)N_ATOMS * F_DIM * sizeof(bf16));
  bf16*  hb   = (bf16*)take((size_t)N_ATOMS * F_DIM * sizeof(bf16));
  float* osum = (float*)take((size_t)N_ATOMS * O_DIM * sizeof(float));

  const int NF  = N_ATOMS * F_DIM;           // 6,400,000
  const int WMAT = F_DIM * F_DIM;            // 16384
  const int gGemm = (N_ATOMS + 127) / 128;   // 391
  const int gElem = (NF + 255) / 256;        // 25000
  const int gEdge = N_EDGES / (8 * 16);      // 6250
  const int gAtomW = N_ATOMS / 8;            // 6250

  // ---- weight conversion / transpose (bf16, n-major)
  for (int m = 0; m < 3; ++m) {
    k_transpose_bf16<<<64, 256, 0, stream>>>(Wj + m * WMAT, wT + (m*9+0)*WMAT, F_DIM, F_DIM);
    k_transpose_bf16<<<64, 256, 0, stream>>>(Wi + m * WMAT, wT + (m*9+1)*WMAT, F_DIM, F_DIM);
    k_transpose_bf16<<<64, 256, 0, stream>>>(Wv + m * WMAT, wT + (m*9+2)*WMAT, F_DIM, F_DIM);
    for (int r = 0; r < 3; ++r)
      for (int l = 0; l < 2; ++l)
        k_transpose_bf16<<<64, 256, 0, stream>>>(resW + (((size_t)m*3 + r)*2 + l)*WMAT,
                                                 wT + (m*9 + 3 + r*2 + l)*WMAT, F_DIM, F_DIM);
    k_transpose_bf16<<<32, 256, 0, stream>>>(G + (size_t)m * K_RBF * F_DIM,
                                             gT + (size_t)m * F_DIM * K_RBF, K_RBF, F_DIM);
  }

  // ---- init
  k_gather_embed<<<gElem, 256, 0, stream>>>(Z, embed, x);
  k_zero<<<(N_ATOMS*O_DIM + 255)/256, 256, 0, stream>>>(osum, N_ATOMS * O_DIM);
  k_rbf<<<N_EDGES / 256, 256, 0, stream>>>(R, esrc, edst, centers, widths, rbfb);

  // ---- interaction blocks
  for (int m = 0; m < 3; ++m) {
    const bf16* WjT = wT + (m*9+0)*WMAT;
    const bf16* WiT = wT + (m*9+1)*WMAT;
    const bf16* WvT = wT + (m*9+2)*WMAT;
    const bf16* GtM = gT + (size_t)m * F_DIM * K_RBF;
    const float* biM = bi + m * F_DIM;
    const float* bjM = bj + m * F_DIM;
    const float* bvM = bv + m * F_DIM;
    const float* uM  = u + m * F_DIM;
    auto resWT = [&](int r, int l) { return wT + (m*9 + 3 + r*2 + l)*WMAT; };
    auto resbP = [&](int r, int l) { return resb + (((size_t)m*3 + r)*2 + l)*F_DIM; };

    // sx = ssp(x); tj = sx@Wj+bj; v = sx@Wi+bi
    k_ssp_bf16<<<gElem, 256, 0, stream>>>(x, sxb, NF);
    k_gemm<0><<<gGemm, 256, 0, stream>>>(sxb, WjT, bjM, nullptr, nullptr, tjb, nullptr, N_ATOMS);
    k_gemm<0><<<gGemm, 256, 0, stream>>>(sxb, WiT, biM, nullptr, nullptr, vb, nullptr, N_ATOMS);
    // v[dst] += (rbf@G) * tj[src]
    k_edge<<<gEdge, 256, 0, stream>>>(rbfb, GtM, tjb, vb, esrc, edst);
    // res_int: h = ssp(ssp(v)@W0+b0); v = v + h@W1+b1
    k_ssp_bf16<<<gElem, 256, 0, stream>>>(vb, sxb, NF);
    k_gemm<1><<<gGemm, 256, 0, stream>>>(sxb, resWT(0,0), resbP(0,0), nullptr, nullptr, nullptr, hb, N_ATOMS);
    k_gemm<2><<<gGemm, 256, 0, stream>>>(hb, resWT(0,1), resbP(0,1), vb, nullptr, vb, nullptr, N_ATOMS);
    // x <- u*x + ssp(v)@Wv + bv
    k_ssp_bf16<<<gElem, 256, 0, stream>>>(vb, sxb, NF);
    k_gemm<3><<<gGemm, 256, 0, stream>>>(sxb, WvT, bvM, x, uM, x, nullptr, N_ATOMS);
    // res_atom on x (in place)
    k_ssp_bf16<<<gElem, 256, 0, stream>>>(x, sxb, NF);
    k_gemm<1><<<gGemm, 256, 0, stream>>>(sxb, resWT(1,0), resbP(1,0), nullptr, nullptr, nullptr, hb, N_ATOMS);
    k_gemm<2><<<gGemm, 256, 0, stream>>>(hb, resWT(1,1), resbP(1,1), x, nullptr, x, nullptr, N_ATOMS);
    // res_out into o
    k_ssp_bf16<<<gElem, 256, 0, stream>>>(x, sxb, NF);
    k_gemm<1><<<gGemm, 256, 0, stream>>>(sxb, resWT(2,0), resbP(2,0), nullptr, nullptr, nullptr, hb, N_ATOMS);
    k_gemm<2><<<gGemm, 256, 0, stream>>>(hb, resWT(2,1), resbP(2,1), x, nullptr, ob, nullptr, N_ATOMS);
    // out_sum += ssp(o)@Wout + bout
    k_out_acc<<<gAtomW, 256, 0, stream>>>(ob, Wout + m * F_DIM * O_DIM, bout + m * O_DIM, osum);
  }

  // ---- molecule reduction
  k_zero<<<(N_MOL * 5 + 255)/256, 256, 0, stream>>>(out, N_MOL * 5);
  k_final<<<(N_ATOMS + 255)/256, 256, 0, stream>>>(osum, Z, scale, shift, amb, R, out);
}
